// TimeKAN_26809185861724
// MI455X (gfx1250) — compile-verified, matrix operand order checked by reading, not run on hardware
//
#include <hip/hip_runtime.h>
#include <math.h>

typedef __attribute__((ext_vector_type(16))) _Float16 v16h;
typedef __attribute__((ext_vector_type(8)))  float    v8f;
typedef __attribute__((ext_vector_type(2)))  float    v2f;

#define DD 128   // channel dim D
#define NB 128   // batch B
#define PI_F 3.14159265358979323846f

// LDS byte offset of a shared-memory pointer (ptrtoint of addrspace(3) ptr).
__device__ inline unsigned lds_offset_of(const void* p) {
    return (unsigned)(unsigned long long)(__attribute__((address_space(3))) const void*)p;
}

// ---------------------------------------------------------------------------
// Build the exact freq_interp matrix G (2L x L):
//   y[m] = sum_n G[m][n] * x[n]   reproduces  irfft(pad(rfft(x)), 2L) * (L/2L)
//   G[m][n] = (1/(4L)) * g[(m-2n) mod 2L]
//   g[0] = L+1 ; g[even j] = (-1)^(j/2) ; g[odd j] = (-1)^((j-1)/2) * cot(pi*j/(2L))
// ---------------------------------------------------------------------------
__global__ void build_G(float* __restrict__ G, int L) {
    int idx = blockIdx.x * blockDim.x + threadIdx.x;
    int total = 2 * L * L;
    if (idx >= total) return;
    int m = idx / L;
    int n = idx - m * L;
    int N2 = 2 * L;
    int j = (m - 2 * n) % N2;
    if (j < 0) j += N2;
    float inv = 1.0f / (4.0f * (float)L);
    float val;
    if (j == 0) {
        val = (float)(L + 1) * inv;
    } else if ((j & 1) == 0) {
        val = (((j >> 1) & 1) ? -1.0f : 1.0f) * inv;
    } else {
        float th = PI_F * (float)j / (2.0f * (float)L);
        float ct = cosf(th) / sinf(th);
        val = ((((j - 1) >> 1) & 1) ? -1.0f : 1.0f) * ct * inv;
    }
    G[idx] = val;
}

// ---------------------------------------------------------------------------
// Transpose/pack coeffs (D,D,deg) f32 -> Bt (D, D*deg) f16 with k = i*deg + d,
// i.e. Bt[o][k] = coeffs[i][o][d].  Matches WMMA B-fragment reads: lane holds
// column o, 16 contiguous k  (= row o of B^T).
// ---------------------------------------------------------------------------
__global__ void build_Bt(const float* __restrict__ C, _Float16* __restrict__ Bt, int deg) {
    int idx = blockIdx.x * blockDim.x + threadIdx.x;
    int K = DD * deg;
    int total = DD * K;
    if (idx >= total) return;
    int o = idx / K;
    int k = idx - o * K;
    int i = k / deg;
    int d = k - i * deg;
    Bt[idx] = (_Float16)C[(i * DD + o) * deg + d];
}

// ---------------------------------------------------------------------------
// Batched upsample GEMM:  Y[b] = Base[b] + sgn * (G (2L x L) @ Xlo[b] (L x D))
//   dec stage:  Base = x_hi, sgn = -1
//   mix stage:  Base = Y (in-place accumulate), sgn = +1
// Block <-> one (batch, 16-row stripe); the 16xL G stripe (contiguous) is
// bulk-staged into LDS with GLOBAL_LOAD_ASYNC_TO_LDS_B128 (ASYNCcnt), then the
// 8 waves (one 16x16 tile each, nt = wave) run V_WMMA_F32_16X16X4_F32 with
// A-fragments from LDS.  Full f32 precision.
// ---------------------------------------------------------------------------
__global__ __launch_bounds__(256) void upsample_gemm(
    const float* __restrict__ G,
    const float* __restrict__ Xlo,   // (B, L, D)
    const float* __restrict__ Base,  // (B, 2L, D)
    float*       __restrict__ Y,     // (B, 2L, D)
    int L, float sgn)
{
    const int N2 = 2 * L;
    __shared__ __attribute__((aligned(16))) float Gs[16 * 256];  // up to 16 KB

    const int tid    = threadIdx.x;
    const int mtiles = N2 / 16;
    const int b  = blockIdx.x / mtiles;
    const int mt = blockIdx.x - b * mtiles;

    // Async-stage the contiguous 16-row G stripe (16*L floats) into LDS.
    {
        const char* src = (const char*)(G + (size_t)mt * 16 * L);
        unsigned ldsbase = lds_offset_of(Gs);
        const int iters = L >> 6;            // (16*L*4) / (256 threads * 16B)
        for (int it = 0; it < iters; ++it) {
            int e = it * 256 + tid;
            unsigned long long gaddr = (unsigned long long)(src + (size_t)e * 16);
            unsigned laddr = ldsbase + e * 16;
            asm volatile("global_load_async_to_lds_b128 %0, %1, off"
                         :: "v"(laddr), "v"(gaddr) : "memory");
        }
        asm volatile("s_wait_asynccnt 0x0" ::: "memory");
    }
    __syncthreads();

    const int nt   = tid >> 5;       // wave -> 16-col channel tile
    const int lane = tid & 31;
    const int lhi  = lane >> 4;      // 0/1 selects K sub-pair
    const int llo  = lane & 15;

    const float* grow = &Gs[llo * L];                               // A row (LDS)
    const float* xcol = Xlo + (size_t)b * L * DD + nt * 16 + llo;   // B col

    v8f c = {};
    for (int kb = 0; kb < L; kb += 4) {
        int k = kb + 2 * lhi;
        v2f a = *(const v2f*)(grow + k);          // Gs[row][k], Gs[row][k+1]
        v2f bb;
        bb.x = xcol[(size_t)k * DD];              // Xlo[b][k][col]
        bb.y = xcol[(size_t)(k + 1) * DD];
        c = __builtin_amdgcn_wmma_f32_16x16x4_f32(false, a, false, bb,
                                                  (short)0, c, false, false);
    }

    #pragma unroll
    for (int r = 0; r < 8; ++r) {
        int m = mt * 16 + r + 8 * lhi;
        size_t idx = ((size_t)b * N2 + m) * DD + nt * 16 + llo;
        Y[idx] = Base[idx] + sgn * c[r];
    }
}

// ---------------------------------------------------------------------------
// Fused m_kan:  Y = ChebyKAN(X) + dw_conv(X)
//   basis T_d(tanh(tanh(x))) built by Chebyshev recurrence into LDS (f16),
//   GEMM [M=B*L, K=D*deg, N=D] via V_WMMA_F32_16X16X32_F16,
//   depthwise 3-tap conv fused into the epilogue.
// One block = 16 output rows x all 128 channels; 8 waves, one 16x16 tile each.
// ---------------------------------------------------------------------------
__global__ __launch_bounds__(256) void cheby_mkan(
    const float*    __restrict__ X,    // (B, L, D) level input
    const _Float16* __restrict__ Bt,   // (D, K) packed coeffs
    const float*    __restrict__ W,    // (D, 3) depthwise weights
    float*          __restrict__ Y,    // (B, L, D) output
    int lgL, int deg)
{
    const int L = 1 << lgL;
    const int K = DD * deg;            // multiple of 32 for deg in {2,3,4,5}
    __shared__ __attribute__((aligned(32))) _Float16 As[16 * 640];

    const int row0 = blockIdx.x * 16;  // first of 16 rows of M = B*L
    const int tid  = threadIdx.x;

    // Phase 1: build f16 Chebyshev basis stripe As[16][K], k = i*deg + d
    for (int e = tid; e < 16 * DD; e += 256) {
        int m = e >> 7;                // row within stripe
        int i = e & (DD - 1);          // input channel
        float x = X[(size_t)(row0 + m) * DD + i];
        float tt = tanhf(tanhf(x));
        _Float16* dst = &As[m * K + i * deg];
        dst[0] = (_Float16)1.0f;
        dst[1] = (_Float16)tt;
        float Tm1 = 1.0f, T = tt;
        for (int d = 2; d < deg; ++d) {
            float Tn = 2.0f * tt * T - Tm1;
            Tm1 = T; T = Tn;
            dst[d] = (_Float16)Tn;
        }
    }
    __syncthreads();

    // Phase 2: each wave computes one 16(M) x 16(N) tile
    const int wave = tid >> 5;         // N-tile 0..7
    const int lane = tid & 31;
    const int lhi  = lane >> 4;
    const int llo  = lane & 15;

    const _Float16* arow = &As[llo * K + lhi * 16];                  // LDS
    const _Float16* brow = Bt + (size_t)(wave * 16 + llo) * K + lhi * 16;

    v8f c = {};
    for (int kb = 0; kb < K; kb += 32) {
        v16h a = *(const v16h*)(arow + kb);   // 16 contiguous k from LDS
        v16h b = *(const v16h*)(brow + kb);   // 16 contiguous k from global
        c = __builtin_amdgcn_wmma_f32_16x16x32_f16(false, a, false, b,
                                                   (short)0, c, false, false);
    }

    // Epilogue: + depthwise conv (kernel 3, zero-padded), store f32
    const int ocol = wave * 16 + llo;
    const float w0 = W[ocol * 3 + 0];
    const float w1 = W[ocol * 3 + 1];
    const float w2 = W[ocol * 3 + 2];
    #pragma unroll
    for (int r = 0; r < 8; ++r) {
        int mrow = row0 + r + 8 * lhi;
        int nn   = mrow & (L - 1);            // time index within batch
        size_t base = (size_t)mrow * DD + ocol;
        float acc = c[r] + w1 * X[base];
        if (nn > 0)     acc += w0 * X[base - DD];
        if (nn < L - 1) acc += w2 * X[base + DD];
        Y[base] = acc;
    }
}

// ---------------------------------------------------------------------------
extern "C" void kernel_launch(void* const* d_in, const int* in_sizes, int n_in,
                              void* d_out, int out_size, void* d_ws, size_t ws_size,
                              hipStream_t stream) {
    (void)in_sizes; (void)n_in; (void)out_size; (void)ws_size;
    const float* x0 = (const float*)d_in[0];   // (128,512,128)
    const float* x1 = (const float*)d_in[1];   // (128,256,128)
    const float* x2 = (const float*)d_in[2];   // (128,128,128)
    const float* x3 = (const float*)d_in[3];   // (128, 64,128)
    const float* c0 = (const float*)d_in[4];  const float* w0 = (const float*)d_in[5];
    const float* c1 = (const float*)d_in[6];  const float* w1 = (const float*)d_in[7];
    const float* c2 = (const float*)d_in[8];  const float* w2 = (const float*)d_in[9];
    const float* c3 = (const float*)d_in[10]; const float* w3 = (const float*)d_in[11];

    float* out = (float*)d_out;                 // (m3_512, m2_256, m1_128, m0_64)
    float* m3 = out;
    float* m2 = m3 + (size_t)NB * 512 * DD;
    float* m1 = m2 + (size_t)NB * 256 * DD;
    float* m0 = m1 + (size_t)NB * 128 * DD;

    // workspace carve (all sizes are multiples of 256 bytes)
    char* p = (char*)d_ws;
    float*    G64  = (float*)p;    p += (size_t)128 * 64  * 4;
    float*    G128 = (float*)p;    p += (size_t)256 * 128 * 4;
    float*    G256 = (float*)p;    p += (size_t)512 * 256 * 4;
    _Float16* Bt0  = (_Float16*)p; p += (size_t)DD * 256 * 2;
    _Float16* Bt1  = (_Float16*)p; p += (size_t)DD * 384 * 2;
    _Float16* Bt2  = (_Float16*)p; p += (size_t)DD * 512 * 2;
    _Float16* Bt3  = (_Float16*)p; p += (size_t)DD * 640 * 2;
    float*    dec2 = (float*)p;    p += (size_t)NB * 128 * DD * 4;
    float*    dec1 = (float*)p;    p += (size_t)NB * 256 * DD * 4;
    float*    dec0 = (float*)p;    p += (size_t)NB * 512 * DD * 4;

    // 1) interpolation matrices
    build_G<<<(2*64*64   + 255)/256, 256, 0, stream>>>(G64,  64);
    build_G<<<(2*128*128 + 255)/256, 256, 0, stream>>>(G128, 128);
    build_G<<<(2*256*256 + 255)/256, 256, 0, stream>>>(G256, 256);

    // 2) packed f16 coefficient matrices
    build_Bt<<<(DD*DD*2 + 255)/256, 256, 0, stream>>>(c0, Bt0, 2);
    build_Bt<<<(DD*DD*3 + 255)/256, 256, 0, stream>>>(c1, Bt1, 3);
    build_Bt<<<(DD*DD*4 + 255)/256, 256, 0, stream>>>(c2, Bt2, 4);
    build_Bt<<<(DD*DD*5 + 255)/256, 256, 0, stream>>>(c3, Bt3, 5);

    // 3) decomposition: dec_i = x_i - up(x_{i+1})      [f32 WMMA + async LDS]
    upsample_gemm<<<NB * (128/16), 256, 0, stream>>>(G64,  x3, x2, dec2,  64, -1.0f);
    upsample_gemm<<<NB * (256/16), 256, 0, stream>>>(G128, x2, x1, dec1, 128, -1.0f);
    upsample_gemm<<<NB * (512/16), 256, 0, stream>>>(G256, x1, x0, dec0, 256, -1.0f);

    // 4) mixing ladder: m0 = m_kan(x3); m_{i+1} = m_kan(dec) + up(m_i)
    cheby_mkan<<<NB *  64 / 16, 256, 0, stream>>>(x3,   Bt0, w0, m0, 6, 2);   // f16 WMMA
    cheby_mkan<<<NB * 128 / 16, 256, 0, stream>>>(dec2, Bt1, w1, m1, 7, 3);
    upsample_gemm<<<NB * (128/16), 256, 0, stream>>>(G64,  m0, m1, m1,  64, 1.0f);
    cheby_mkan<<<NB * 256 / 16, 256, 0, stream>>>(dec1, Bt2, w2, m2, 8, 4);
    upsample_gemm<<<NB * (256/16), 256, 0, stream>>>(G128, m1, m2, m2, 128, 1.0f);
    cheby_mkan<<<NB * 512 / 16, 256, 0, stream>>>(dec0, Bt3, w3, m3, 9, 5);
    upsample_gemm<<<NB * (512/16), 256, 0, stream>>>(G256, m2, m3, m3, 256, 1.0f);
}
